// CorrelationEncoder_4939212390930
// MI455X (gfx1250) — matrix operand order repaired and checked by reading, not compiled, not run on hardware
//
#include <hip/hip_runtime.h>

// CorrelationEncoder forward for MI455X (gfx1250, wave32).
// All dense contractions run through one generic bf16-WMMA tiled GEMM
// (v_wmma_f32_16x16x32_bf16, fp32 accumulate). Requires ws_size >= ~299MB.

typedef __attribute__((ext_vector_type(8)))  float  v8f;
typedef __attribute__((ext_vector_type(16))) __bf16 v16bf;
typedef __attribute__((ext_vector_type(4)))  __bf16 v4bf;
typedef __attribute__((ext_vector_type(2)))  __bf16 v2bf;

// ------------------------------------------------------------------
// Generic tiled WMMA GEMM:  C[m,n] = act( beta*C + bias + sum_k A(m,k)*B(k,n) )
// A(m,k) addr = (m/GA)*sAg + (m%GA)*sAr + k*sAk. NOTE: sAk==1 implies plain
// A (GA unused), which enables the vectorized full-tile fast path.
// B(k,n) addr = k*sBk + n*sBn ;  C plain row-major with ldc. blockIdx.z batches.
// ------------------------------------------------------------------
struct GemmArgs {
  int M, N, K, GA;
  long sAg, sAr, sAk, zA;
  long sBk, sBn, zB;
  long ldc, zC;
  const float* A; const float* B; float* C; const float* bias;
};

// ACT: 0 none, 1 relu, 2 gelu(erf). BIASM: 0 none, 1 per-n, 2 per-m.
// BETA: 0 overwrite, 1 accumulate into C. AACT: 0 none, 1 relu on A load.
template<int ACT, int BIASM, int BETA, int AACT>
__global__ __launch_bounds__(128) void gemm_wmma_k(GemmArgs g) {
  __shared__ __bf16 As[64 * 40];   // [m][k], padded stride
  __shared__ __bf16 Bs[64 * 40];   // [n][k] (transposed in LDS)
  const int tid = threadIdx.x, lane = tid & 31, wid = tid >> 5;
  const int wr = wid >> 1, wc = wid & 1, l16 = lane & 15, hi = lane >> 4;
  const long m0 = (long)blockIdx.y * 64, n0 = (long)blockIdx.x * 64;
  const float* Ab = g.A + (long)blockIdx.z * g.zA;
  const float* Bb = g.B + (long)blockIdx.z * g.zB;
  float*       Cb = g.C + (long)blockIdx.z * g.zC;
  const bool fullM = (m0 + 64 <= g.M);
  const bool fullN = (n0 + 64 <= g.N);
  v8f acc[2][2] = {};

  for (int k0 = 0; k0 < g.K; k0 += 32) {
    const bool fullK = (k0 + 32 <= g.K);
    // ---------------- stage A (fp32 -> bf16) ----------------
    if (g.sAk == 1 && fullM && fullK) {
      // full tile, plain A: 16B vector loads, packed bf16, 8B LDS stores
#pragma unroll
      for (int it = 0; it < 4; ++it) {
        int idx = it * 128 + tid;
        int r = idx >> 3, kq = (idx & 7) << 2;
        const float4 x = *(const float4*)(Ab + (m0 + r) * g.sAr + (k0 + kq));
        float a0 = x.x, a1 = x.y, a2 = x.z, a3 = x.w;
        if (AACT == 1) {
          a0 = fmaxf(a0, 0.f); a1 = fmaxf(a1, 0.f);
          a2 = fmaxf(a2, 0.f); a3 = fmaxf(a3, 0.f);
        }
        v4bf h; h[0] = (__bf16)a0; h[1] = (__bf16)a1; h[2] = (__bf16)a2; h[3] = (__bf16)a3;
        *(v4bf*)&As[r * 40 + kq] = h;
      }
      // speculative prefetch of next K slab, one line per row (global_prefetch_b8)
      if (k0 + 32 < g.K && tid < 64)
        __builtin_prefetch(Ab + (m0 + tid) * g.sAr + (k0 + 32), 0, 1);
    } else if (g.sAk == 1) {
      // partial tile, plain A: predicated scalar (k-fast, coalesced)
      for (int idx = tid; idx < 64 * 32; idx += 128) {
        int r = idx >> 5, kk = idx & 31;
        int m = (int)m0 + r, k = k0 + kk;
        float x = 0.f;
        if (m < g.M && k < g.K) {
          x = Ab[(long)m * g.sAr + (long)k];
          if (AACT == 1) x = fmaxf(x, 0.f);
        }
        As[r * 40 + kk] = (__bf16)x;
      }
    } else {
      // grouped/strided A (sp_lin chunks, freq_lin): r-fast (sAr==1 coalesced)
      for (int idx = tid; idx < 64 * 32; idx += 128) {
        int r = idx & 63, kk = idx >> 6;
        int m = (int)m0 + r, k = k0 + kk;
        float x = 0.f;
        if (m < g.M && k < g.K) {
          x = Ab[(long)(m / g.GA) * g.sAg + (long)(m % g.GA) * g.sAr + (long)k * g.sAk];
          if (AACT == 1) x = fmaxf(x, 0.f);
        }
        As[r * 40 + kk] = (__bf16)x;
      }
    }
    // ---------------- stage B transposed: Bs[n][k] ----------------
    if (fullN && fullK) {
#pragma unroll
      for (int it = 0; it < 8; ++it) {
        int idx = it * 128 + tid;
        int nl = idx & 63, kp = idx >> 6;              // pair of k values
        const float* src = Bb + (long)(k0 + 2 * kp) * g.sBk + (long)(n0 + nl) * g.sBn;
        v2bf h; h[0] = (__bf16)src[0]; h[1] = (__bf16)src[g.sBk];
        *(v2bf*)&Bs[nl * 40 + 2 * kp] = h;
      }
    } else {
      for (int idx = tid; idx < 32 * 64; idx += 128) {
        int nl = idx & 63, kk = idx >> 6;
        int n = (int)n0 + nl, k = k0 + kk;
        float x = 0.f;
        if (n < g.N && k < g.K) x = Bb[(long)k * g.sBk + (long)n * g.sBn];
        Bs[nl * 40 + kk] = (__bf16)x;
      }
    }
    __syncthreads();

    // ---- fragments per ISA layouts (wave32) ----
    v16bf af[2], bfr[2];
#pragma unroll
    for (int mi = 0; mi < 2; ++mi) {
      int row = wr * 32 + mi * 16 + l16, kb = hi ? 8 : 0;
#pragma unroll
      for (int e = 0; e < 8; ++e) {
        af[mi][e]     = As[row * 40 + kb + e];        // K 0..7 / 8..15
        af[mi][8 + e] = As[row * 40 + kb + 16 + e];   // K 16..23 / 24..31
      }
    }
#pragma unroll
    for (int nj = 0; nj < 2; ++nj) {
      int col = wc * 32 + nj * 16 + l16, kb = hi ? 8 : 0;
#pragma unroll
      for (int e = 0; e < 8; ++e) {
        bfr[nj][e]     = Bs[col * 40 + kb + e];
        bfr[nj][8 + e] = Bs[col * 40 + kb + 16 + e];
      }
    }
#pragma unroll
    for (int mi = 0; mi < 2; ++mi)
#pragma unroll
      for (int nj = 0; nj < 2; ++nj)
        acc[mi][nj] = __builtin_amdgcn_wmma_f32_16x16x32_bf16(
            false, af[mi], false, bfr[nj], (short)0, acc[mi][nj], false, false);
    __syncthreads();
  }

  // ---- epilogue (C layout: lanes 0-15 M=r, lanes 16-31 M=r+8; N=lane%16) ----
  if (fullM && fullN) {
#pragma unroll
    for (int mi = 0; mi < 2; ++mi)
      for (int nj = 0; nj < 2; ++nj)
#pragma unroll
        for (int r = 0; r < 8; ++r) {
          long m = m0 + wr * 32 + mi * 16 + r + hi * 8;
          long n = n0 + wc * 32 + nj * 16 + l16;
          float x = acc[mi][nj][r];
          if (BIASM == 1) x += g.bias[n];
          if (BIASM == 2) x += g.bias[m];
          if (BETA == 1)  x += Cb[m * g.ldc + n];
          if (ACT == 1) x = fmaxf(x, 0.f);
          if (ACT == 2) x = 0.5f * x * (1.f + erff(x * 0.70710678118655f));
          Cb[m * g.ldc + n] = x;
        }
  } else {
#pragma unroll
    for (int mi = 0; mi < 2; ++mi)
      for (int nj = 0; nj < 2; ++nj)
#pragma unroll
        for (int r = 0; r < 8; ++r) {
          long m = m0 + wr * 32 + mi * 16 + r + hi * 8;
          long n = n0 + wc * 32 + nj * 16 + l16;
          if (m < g.M && n < g.N) {
            float x = acc[mi][nj][r];
            if (BIASM == 1) x += g.bias[n];
            if (BIASM == 2) x += g.bias[m];
            if (BETA == 1)  x += Cb[m * g.ldc + n];
            if (ACT == 1) x = fmaxf(x, 0.f);
            if (ACT == 2) x = 0.5f * x * (1.f + erff(x * 0.70710678118655f));
            Cb[m * g.ldc + n] = x;
          }
        }
  }
}

// ------------------------------------------------------------------
// Elementwise / small kernels
// ------------------------------------------------------------------
__global__ void add_pos_k(const float* x, const float* pos, float* y, long n) {
  long i = (long)blockIdx.x * blockDim.x + threadIdx.x;
  if (i < n) y[i] = x[i] + pos[i % 1536];
}

// x:(n,v,l,t,s,c) -> xt tokens ((n,v),(l,s),c) for fixed t. n = 4000*24*32
__global__ void gather_t_k(const float* x, float* xt, int t, long n) {
  long i = (long)blockIdx.x * blockDim.x + threadIdx.x;
  if (i >= n) return;
  int c = (int)(i & 31); long r = i >> 5; int ls = (int)(r % 24); long bb = r / 24;
  int l = ls >> 1, s = ls & 1;
  xt[i] = x[bb * 1536 + (long)((l * 4 + t * 2 + s) * 32 + c)];
}

__global__ void gate_combine_k(const float* f, const float* g, float* xg, int t, long n) {
  long i = (long)blockIdx.x * blockDim.x + threadIdx.x;
  if (i >= n) return;
  int c = (int)(i & 31); long r = i >> 5; int ls = (int)(r % 24); long bb = r / 24;
  int l = ls >> 1, s = ls & 1;
  float val = tanhf(f[i]) * (1.f / (1.f + expf(-g[i])));
  xg[bb * 1536 + (long)((l * 4 + t * 2 + s) * 32 + c)] = val;
}

// h0[s][nt][c][l][v] <- xg (n,v,l,t,s,c); nt = n*2+t
__global__ void build_h0_k(const float* xg, float* h0, long n) {
  long i = (long)blockIdx.x * blockDim.x + threadIdx.x;
  if (i >= n) return;
  int v = (int)(i % 500); long r = i / 500;
  int l = (int)(r % 12);  r /= 12;
  int c = (int)(r % 32);  r /= 32;
  int nt = (int)(r % 16); int s = (int)(r / 16);
  int nn = nt >> 1, t = nt & 1;
  long bb = (long)nn * 500 + v;
  h0[i] = xg[bb * 1536 + (long)((l * 4 + t * 2 + s) * 32 + c)];
}

// x[i] = BN( spout[s][nt][l][v][c] + x[i] ), channel = (t,s,c) = i%128
__global__ void bn_rearrange_k(const float* spout, float* x, const float* w,
                               const float* b, const float* m, const float* vv, long n) {
  long i = (long)blockIdx.x * blockDim.x + threadIdx.x;
  if (i >= n) return;
  int c = (int)(i & 31);
  int s = (int)((i >> 5) & 1);
  int t = (int)((i >> 6) & 1);
  long r = i >> 7;
  int l = (int)(r % 12); r /= 12;
  int v = (int)(r % 500); long nn = r / 500;
  long sp = ((((long)s * 16 + nn * 2 + t) * 12 + l) * 500 + v) * 32 + c;
  int ch = (int)(i & 127);
  float val = spout[sp] + x[i];
  x[i] = (val - m[ch]) * rsqrtf(vv[ch] + 1e-5f) * w[ch] + b[ch];
}

// pack wq|wk|wv (each 32x32) -> wp (32x96); biases -> bp (96)
__global__ void pack_qkv_k(const float* wq, const float* wk, const float* wv,
                           const float* bq, const float* bk, const float* bv,
                           float* wp, float* bp) {
  int i = blockIdx.x * blockDim.x + threadIdx.x;
  if (i < 1024) {
    int k = i >> 5, c = i & 31;
    wp[k * 96 + c]      = wq[i];
    wp[k * 96 + 32 + c] = wk[i];
    wp[k * 96 + 64 + c] = wv[i];
  } else if (i < 1056) {
    int c = i - 1024;
    bp[c] = bq[c]; bp[32 + c] = bk[c]; bp[64 + c] = bv[c];
  }
}

// LayerNorm over last dim 32; one wave per row; y = LN(a+res)*g + b
__global__ void ln32_k(const float* a, const float* res, const float* g,
                       const float* b, float* out, long rows) {
  long row = (long)blockIdx.x * (blockDim.x >> 5) + (threadIdx.x >> 5);
  if (row >= rows) return;
  int lane = threadIdx.x & 31;
  float x = a[row * 32 + lane] + (res ? res[row * 32 + lane] : 0.f);
  float mu = x;
  for (int o = 16; o; o >>= 1) mu += __shfl_xor(mu, o);
  mu *= (1.f / 32.f);
  float d = x - mu, var = d * d;
  for (int o = 16; o; o >>= 1) var += __shfl_xor(var, o);
  var *= (1.f / 32.f);
  out[row * 32 + lane] = d * rsqrtf(var + 1e-5f) * g[lane] + b[lane];
}

// Per-(batch,head) attention; qkv interleaved tokens (B,S,96): q|k|v per token.
template<int S>
__global__ void attn_k(const float* qkv, float* out) {
  int bh = blockIdx.x; int b = bh >> 2; int h = bh & 3;
  int lane = threadIdx.x;
  if (lane >= S) return;
  const float scale = 0.35355339059327f;  // 1/sqrt(8)
  long tb = (long)b * S;
  const float* qp = qkv + (tb + lane) * 96 + h * 8;
  float qr[8];
#pragma unroll
  for (int d = 0; d < 8; ++d) qr[d] = qp[d];
  float sc[S]; float mx = -1e30f;
  for (int j = 0; j < S; ++j) {
    const float* kp = qkv + (tb + j) * 96 + 32 + h * 8;
    float a = 0.f;
#pragma unroll
    for (int d = 0; d < 8; ++d) a += qr[d] * kp[d];
    a *= scale; sc[j] = a; mx = fmaxf(mx, a);
  }
  float sum = 0.f;
  for (int j = 0; j < S; ++j) { sc[j] = expf(sc[j] - mx); sum += sc[j]; }
  float inv = 1.f / sum;
  float o[8] = {0, 0, 0, 0, 0, 0, 0, 0};
  for (int j = 0; j < S; ++j) {
    const float* vp = qkv + (tb + j) * 96 + 64 + h * 8;
    float pj = sc[j] * inv;
#pragma unroll
    for (int d = 0; d < 8; ++d) o[d] += pj * vp[d];
  }
  float* op = out + (tb + lane) * 32 + h * 8;
#pragma unroll
  for (int d = 0; d < 8; ++d) op[d] = o[d];
}

// out[v,w] = (A(v,w) + I) / colsum ; transpose: A(v,w) = a[w,v]. One wave/col.
__global__ void colnorm_sym_k(const float* a, float* out, int transpose) {
  int col = blockIdx.x * 4 + (threadIdx.x >> 5);
  int lane = threadIdx.x & 31;
  if (col >= 500) return;
  float s = 0.f;
  for (int v = lane; v < 500; v += 32) {
    float x = transpose ? a[(long)col * 500 + v] : a[(long)v * 500 + col];
    if (v == col) x += 1.f;
    s += x;
  }
  for (int o = 16; o; o >>= 1) s += __shfl_xor(s, o);
  float inv = 1.f / s;
  for (int v = lane; v < 500; v += 32) {
    float x = transpose ? a[(long)col * 500 + v] : a[(long)v * 500 + col];
    if (v == col) x += 1.f;
    out[(long)v * 500 + col] = x * inv;
  }
}

// softmax over axis v (rows) of (s,v,w); one wave per (s,w) column
__global__ void col_softmax_k(float* a, int V, int ncols) {
  int col = blockIdx.x * 4 + (threadIdx.x >> 5);
  int lane = threadIdx.x & 31;
  if (col >= ncols) return;
  float* base = a + (long)(col / V) * V * V + (col % V);
  float mx = -1e30f;
  for (int v = lane; v < V; v += 32) mx = fmaxf(mx, base[(long)v * V]);
  for (int o = 16; o; o >>= 1) mx = fmaxf(mx, __shfl_xor(mx, o));
  float sum = 0.f;
  for (int v = lane; v < V; v += 32) sum += expf(base[(long)v * V] - mx);
  for (int o = 16; o; o >>= 1) sum += __shfl_xor(sum, o);
  float inv = 1.f / sum;
  for (int v = lane; v < V; v += 32) base[(long)v * V] = expf(base[(long)v * V] - mx) * inv;
}

// ------------------------------------------------------------------
// Host-side orchestration
// ------------------------------------------------------------------
enum { SEL_BIASN_NONE = 0, SEL_BIASN_GELU = 1, SEL_RELU_NOBIAS = 2, SEL_MLP = 3,
       SEL_PLAIN = 4, SEL_ACC = 5, SEL_ACC_BIASN = 6, SEL_BIASM = 7 };

static void launch_gemm(hipStream_t st, int sel, const GemmArgs& ga, int Z) {
  dim3 gr((ga.N + 63) / 64, (ga.M + 63) / 64, Z), bl(128);
  switch (sel) {
    case SEL_BIASN_NONE: gemm_wmma_k<0, 1, 0, 0><<<gr, bl, 0, st>>>(ga); break;
    case SEL_BIASN_GELU: gemm_wmma_k<2, 1, 0, 0><<<gr, bl, 0, st>>>(ga); break;
    case SEL_RELU_NOBIAS:gemm_wmma_k<1, 0, 0, 0><<<gr, bl, 0, st>>>(ga); break;
    case SEL_MLP:        gemm_wmma_k<1, 1, 0, 1><<<gr, bl, 0, st>>>(ga); break;
    case SEL_PLAIN:      gemm_wmma_k<0, 0, 0, 0><<<gr, bl, 0, st>>>(ga); break;
    case SEL_ACC:        gemm_wmma_k<0, 0, 1, 0><<<gr, bl, 0, st>>>(ga); break;
    case SEL_ACC_BIASN:  gemm_wmma_k<0, 1, 1, 0><<<gr, bl, 0, st>>>(ga); break;
    case SEL_BIASM:      gemm_wmma_k<0, 2, 0, 0><<<gr, bl, 0, st>>>(ga); break;
  }
}

static void gemm_general(hipStream_t st, int sel, int M, int N, int K,
                         const float* A, int GA, long sAg, long sAr, long sAk, long zA,
                         const float* B, long sBk, long sBn, long zB,
                         float* C, long ldc, long zC, const float* bias, int Z) {
  GemmArgs ga; ga.M = M; ga.N = N; ga.K = K; ga.GA = GA;
  ga.sAg = sAg; ga.sAr = sAr; ga.sAk = sAk; ga.zA = zA;
  ga.sBk = sBk; ga.sBn = sBn; ga.zB = zB;
  ga.ldc = ldc; ga.zC = zC; ga.A = A; ga.B = B; ga.C = C; ga.bias = bias;
  launch_gemm(st, sel, ga, Z);
}

static void gemm_plainA(hipStream_t st, int sel, int M, int N, int K,
                        const float* A, long lda, const float* B, long sBk, long sBn,
                        float* C, long ldc, const float* bias) {
  gemm_general(st, sel, M, N, K, A, 1 << 30, 0, lda, 1, 0, B, sBk, sBn, 0, C, ldc, 0, bias, 1);
}

struct Mha {
  const float *wq, *bq, *wk, *bk, *wv, *bv, *wf, *bf, *g, *b, *w1, *b1, *w2, *b2;
};
struct LayerP {
  Mha fil[2], gate[2];
  const float *emb1, *emb2, *spw, *spb, *skw, *skb, *bnw, *bnb, *bnm, *bnv;
};

// qkvff: >= T*128 floats (holds qkv, later reused for the 128-wide FF).
static void run_mha(hipStream_t st, const Mha& p, const float* xin, float* out,
                    int B, int S, float* qkvff, float* r, float* r2,
                    float* wp, float* bp) {
  int T = B * S;
  pack_qkv_k<<<9, 128, 0, st>>>(p.wq, p.wk, p.wv, p.bq, p.bk, p.bv, wp, bp);
  gemm_plainA(st, SEL_BIASN_NONE, T, 96, 32, xin, 32, wp, 96, 1, qkvff, 96, bp);
  if (S == 24) attn_k<24><<<B * 4, 32, 0, st>>>(qkvff, r);
  else         attn_k<4><<<B * 4, 32, 0, st>>>(qkvff, r);
  gemm_plainA(st, SEL_BIASN_NONE, T, 32, 32, r, 32, p.wf, 32, 1, r2, 32, p.bf);
  ln32_k<<<(T + 7) / 8, 256, 0, st>>>(r2, xin, p.g, p.b, r, (long)T);
  gemm_plainA(st, SEL_BIASN_GELU, T, 128, 32, r, 32, p.w1, 128, 1, qkvff, 128, p.b1);
  gemm_plainA(st, SEL_BIASN_NONE, T, 32, 128, qkvff, 128, p.w2, 32, 1, r2, 32, p.b2);
  ln32_k<<<(T + 7) / 8, 256, 0, st>>>(r2, r, p.g, p.b, out, (long)T);
}

extern "C" void kernel_launch(void* const* d_in, const int* in_sizes, int n_in,
                              void* d_out, int out_size, void* d_ws, size_t ws_size,
                              hipStream_t stream) {
  (void)in_sizes; (void)n_in; (void)out_size; (void)ws_size;
  const float* x_in = (const float*)d_in[0];
  const float* supports = (const float*)d_in[1];
  int ip = 2;
  auto nextT = [&]() { return (const float*)d_in[ip++]; };
  auto readMha = [&]() {
    Mha m;
    m.wq = nextT(); m.bq = nextT(); m.wk = nextT(); m.bk = nextT();
    m.wv = nextT(); m.bv = nextT(); m.wf = nextT(); m.bf = nextT();
    m.g = nextT();  m.b = nextT();
    m.w1 = nextT(); m.b1 = nextT(); m.w2 = nextT(); m.b2 = nextT();
    return m;
  };
  const float* pos = nextT();
  LayerP LP[2];
  for (int l = 0; l < 2; ++l) {
    for (int t = 0; t < 2; ++t) LP[l].fil[t] = readMha();
    for (int t = 0; t < 2; ++t) LP[l].gate[t] = readMha();
    LP[l].emb1 = nextT(); LP[l].emb2 = nextT();
    LP[l].spw = nextT();  LP[l].spb = nextT();
    LP[l].skw = nextT();  LP[l].skb = nextT();
    LP[l].bnw = nextT();  LP[l].bnb = nextT();
    LP[l].bnm = nextT();  LP[l].bnv = nextT();
  }
  const float* mlpw = nextT(); const float* mlpb = nextT();
  const float* outw = nextT(); const float* outb = nextT();
  Mha freq = readMha();
  const float* flw = nextT(); const float* flb = nextT();

  // ---- workspace layout (floats); total ~74.7M floats (~299MB) ----
  float* W = (float*)d_ws;
  float* xbuf  = W;                    // 6,144,000  x / residual
  float* xg    = W + 6144000L;         // 6,144,000  gated x
  float* skip  = W + 12288000L;        // 6,144,000  skip accumulator
  float* sup   = W + 18432000L;        // 1,000,000  4 normalized supports
  float* filt  = W + 19432000L;        // 3,072,000
  float* gatet = W + 22504000L;        // 3,072,000
  float* fbuf  = filt;                 // reuse (6,144,000) for freq tokens
  float* PXT = W + 25576000L;          // 6,144,000  xt / h0 / obuf
  float* PQ  = PXT + 6144000L;         // 6,144,000  r / h1 / hbuf
  float* PK  = PQ  + 6144000L;         // 6,144,000  r2 / spout
  float* PV  = PK  + 6144000L;         // 6,144,000  adp
  float* PFF = PV  + 6144000L;         // 24,576,000 qkv (T*96) / ff (T*128)
  float* wpk = PFF + 24576000L;        // 3,072 packed qkv weights
  float* bpk = wpk + 3072;             // 96 packed qkv bias

  hipMemsetAsync(skip, 0, 6144000L * sizeof(float), stream);
  add_pos_k<<<(6144000 + 255) / 256, 256, 0, stream>>>(x_in, pos, xbuf, 6144000L);

  // normalized supports: [s0, s0^T, s1, s1^T]
  for (int i = 0; i < 2; ++i) {
    colnorm_sym_k<<<125, 128, 0, stream>>>(supports + (long)i * 250000, sup + (long)(2 * i) * 250000, 0);
    colnorm_sym_k<<<125, 128, 0, stream>>>(supports + (long)i * 250000, sup + (long)(2 * i + 1) * 250000, 1);
  }

  for (int l = 0; l < 2; ++l) {
    // ---- temporal (per t: fil & gate MHA over tokens (n,v)x(l,s)) ----
    for (int t = 0; t < 2; ++t) {
      gather_t_k<<<(3072000 + 255) / 256, 256, 0, stream>>>(xbuf, PXT, t, 3072000L);
      run_mha(stream, LP[l].fil[t],  PXT, filt,  4000, 24, PFF, PQ, PK, wpk, bpk);
      run_mha(stream, LP[l].gate[t], PXT, gatet, 4000, 24, PFF, PQ, PK, wpk, bpk);
      gate_combine_k<<<(3072000 + 255) / 256, 256, 0, stream>>>(filt, gatet, xg, t, 3072000L);
    }
    // skip += xg @ skw + skb
    gemm_general(stream, SEL_ACC_BIASN, 192000, 32, 32, xg, 1 << 30, 0, 32, 1, 0,
                 LP[l].skw, 32, 1, 0, skip, 32, 0, LP[l].skb, 1);
    // adaptive support: softmax_v(relu(emb1 @ emb2)), batched over s via grid.z
    float* adp = PV;
    gemm_general(stream, SEL_RELU_NOBIAS, 500, 500, 16,
                 LP[l].emb1, 1 << 30, 0, 16, 1, 8000,
                 LP[l].emb2, 500, 1, 8000,
                 adp, 500, 250000, nullptr, 2);
    col_softmax_k<<<250, 128, 0, stream>>>(adp, 500, 1000);
    // spatial: h0[s][nt][c][l][v]; conv GEMMs over v; sp_lin accumulated chunk-wise
    float* h0 = PXT; float* h1 = PQ; float* spout = PK;
    build_h0_k<<<(6144000 + 255) / 256, 256, 0, stream>>>(xg, h0, 6144000L);
    // chunk 0 = h0: write + bias (batched over s)
    gemm_general(stream, SEL_BIASN_NONE, 96000, 32, 32,
                 h0, 6000, 192000, 1, 6000, 3072000,
                 LP[l].spw, 32, 1, 0, spout, 32, 3072000, LP[l].spb, 2);
    float* hp = h0; float* hn = h1;
    for (int app = 0; app < 10; ++app) {
      const float* B2 = (app < 2) ? adp : (sup + (long)((app - 2) / 2) * 250000);
      long zB2 = (app < 2) ? 250000 : 0;
      gemm_general(stream, SEL_PLAIN, 6144, 500, 500,
                   hp, 1 << 30, 0, 500, 1, 3072000,
                   B2, 500, 1, zB2,
                   hn, 500, 3072000, nullptr, 2);
      const float* wj = LP[l].spw + (long)(app + 1) * 32 * 32;
      gemm_general(stream, SEL_ACC, 96000, 32, 32,
                   hn, 6000, 192000, 1, 6000, 3072000,
                   wj, 32, 1, 0, spout, 32, 3072000, nullptr, 2);
      float* tmp = hp; hp = hn; hn = tmp;
    }
    bn_rearrange_k<<<(6144000 + 255) / 256, 256, 0, stream>>>(
        spout, xbuf, LP[l].bnw, LP[l].bnb, LP[l].bnm, LP[l].bnv, 6144000L);
  }

  // ---- output head ----
  float* hbuf = PQ; float* obuf = PXT;
  gemm_plainA(stream, SEL_MLP, 48000, 128, 128, skip, 128, mlpw, 128, 1, hbuf, 128, mlpb);
  gemm_plainA(stream, SEL_BIASN_NONE, 48000, 128, 128, hbuf, 128, outw, 128, 1, obuf, 128, outb);
  // freq attention over tokens (n,v,l) x (t,s), Dm=32
  run_mha(stream, freq, obuf, fbuf, 48000, 4, PFF, PQ, PK, wpk, bpk);
  // freq_lin: out[b,o,c] = sum_k W[k,o] f[b,k,c] + bias[o]; A = W^T, batched z=b
  gemm_general(stream, SEL_BIASM, 12, 32, 48,
               flw, 1 << 30, 0, 1, 12, 0,
               fbuf, 32, 1, 1536,
               (float*)d_out, 32, 384, flb, 4000);
}
